// Attention4D_72215580115669
// MI455X (gfx1250) — compile-verified, matrix operand order checked by reading, not run on hardware
//
#include <hip/hip_runtime.h>

// ---------------------------------------------------------------------------
// Attention4D (LeViT-style) fused kernel for MI455X (gfx1250, wave32, WMMA).
//
// Roofline: ~210 GFLOP, ~310 MB min HBM traffic (13 us @ 23.3 TB/s) when
// fused -> compute bound on the f16 WMMA pipe. One workgroup per batch row,
// all intermediates (qkv, attention, P, O) live in LDS (~315 KB of 320 KB).
// ---------------------------------------------------------------------------

typedef __attribute__((ext_vector_type(16))) _Float16 v16h;
typedef __attribute__((ext_vector_type(4)))  _Float16 v4h;
typedef __attribute__((ext_vector_type(8)))  float    v8f;

#define NUM_HEADS 8
#define KEY_DIM   32
#define VAL_DIM   128
#define NTOK      49
#define RES       7
#define DMODEL    384
#define QKV_OUT   1536          // 8 * (32 + 32 + 128)
#define HSTRIDE   192           // per-head slice of qkv row
#define MPAD      64            // 49 tokens padded to 64 for 16x16 tiles
#define KSTEPS_QKV 12           // 384 / 32
#define KSTEPS_PROJ 32          // 1024 / 32

struct SharedMem {
    _Float16 qk[MPAD][NUM_HEADS * 64];          // [row][h*64 + d] d<32: Q, else K  (64 KB)
    _Float16 vts[NUM_HEADS][VAL_DIM][MPAD];     // V transposed [h][n][k]          (128 KB)
    float    absm[NUM_HEADS][NTOK + 1];         // attention bias table           (1.6 KB)
    union {
        _Float16 xs[MPAD][DMODEL];              // stage A/B input                  (48 KB)
        struct {
            _Float16 o[NTOK][NUM_HEADS * VAL_DIM]; // attention output              (98 KB)
            _Float16 p[2][MPAD][MPAD];             // softmax probs, 2 heads        (16 KB)
        } c;
    } u;
};                                              // total ~315 KB <= 320 KB

// --- tiny one-shot kernel: f32 weights -> f16, transposed to [n][k] ---------
__global__ void convert_weights(const float* __restrict__ qkvK,
                                const float* __restrict__ projK,
                                _Float16* __restrict__ qkvT,
                                _Float16* __restrict__ projT) {
    int idx = blockIdx.x * blockDim.x + threadIdx.x;
    int stride = gridDim.x * blockDim.x;
    for (int t = idx; t < QKV_OUT * DMODEL; t += stride) {
        int n = t / DMODEL, k = t % DMODEL;
        qkvT[t] = (_Float16)qkvK[k * QKV_OUT + n];
    }
    for (int t = idx; t < DMODEL * 1024; t += stride) {
        int n = t / 1024, k = t % 1024;
        projT[t] = (_Float16)projK[k * DMODEL + n];
    }
}

__global__ __launch_bounds__(256)
void attn4d_fused(const float* __restrict__ x,
                  const _Float16* __restrict__ qkvT,
                  const float* __restrict__ qkv_bias,
                  const _Float16* __restrict__ projT,
                  const float* __restrict__ proj_bias,
                  const float* __restrict__ ab,
                  float* __restrict__ out) {
    extern __shared__ char smem_raw[];
    SharedMem& sm = *reinterpret_cast<SharedMem*>(smem_raw);

    const int tid     = threadIdx.x;
    const int b       = blockIdx.x;
    const int w       = tid >> 5;       // wave id 0..7
    const int lane    = tid & 31;
    const int laneN   = lane & 15;      // column within 16x16 tile
    const int halfSel = lane >> 4;      // which 16-element K half this lane owns

    // ---- Stage A: x_b (49x384 f32) -> f16 LDS, vectorized b128 loads -------
    const float4* xb4 = reinterpret_cast<const float4*>(x + (size_t)b * NTOK * DMODEL);
    for (int t = tid; t < MPAD * (DMODEL / 4); t += 256) {
        int r = t / (DMODEL / 4), c4 = t % (DMODEL / 4);
        v4h hv;
        if (r < NTOK) {
            float4 f = xb4[r * (DMODEL / 4) + c4];
            hv[0] = (_Float16)f.x; hv[1] = (_Float16)f.y;
            hv[2] = (_Float16)f.z; hv[3] = (_Float16)f.w;
        } else {
            hv[0] = hv[1] = hv[2] = hv[3] = (_Float16)0.f;
        }
        *(v4h*)&sm.u.xs[r][c4 * 4] = hv;
    }
    // stage attention-bias table into LDS (reused 8x per block from LDS)
    for (int t = tid; t < NUM_HEADS * NTOK; t += 256)
        sm.absm[t / NTOK][t % NTOK] = ab[t];
    __syncthreads();

    // ---- Stage B: qkv = x @ Wqkv + bias  (M=64, N=1536, K=384) -------------
    // 96 n-tiles over 8 waves; B-fragments hoisted across the 4 m-tiles.
    for (int nt = w; nt < QKV_OUT / 16; nt += 8) {
        const _Float16* wc = qkvT + (size_t)(nt * 16 + laneN) * DMODEL + halfSel * 16;
        // warm L0/L2 for this wave's next n-tile (global_prefetch_b8)
        __builtin_prefetch(wc + (size_t)8 * 16 * DMODEL, 0, 0);
        v16h bfrag[KSTEPS_QKV];
#pragma unroll
        for (int k = 0; k < KSTEPS_QKV; ++k)
            bfrag[k] = *(const v16h*)(wc + k * 32);
        const float bias = qkv_bias[nt * 16 + laneN];
        const int cg = nt * 16 + laneN;
        const int h  = cg / HSTRIDE;
        const int rr = cg % HSTRIDE;
        for (int mt = 0; mt < 4; ++mt) {
            v8f acc;
#pragma unroll
            for (int i = 0; i < 8; ++i) acc[i] = bias;
#pragma unroll
            for (int k = 0; k < KSTEPS_QKV; ++k) {
                v16h afrag = *(const v16h*)&sm.u.xs[mt * 16 + laneN][k * 32 + halfSel * 16];
                acc = __builtin_amdgcn_wmma_f32_16x16x32_f16(
                    false, afrag, false, bfrag[k], (short)0, acc, false, false);
            }
            // scatter C tile: Q/K row-major per head, V transposed
#pragma unroll
            for (int r = 0; r < 8; ++r) {
                int row = mt * 16 + halfSel * 8 + r;
                _Float16 hv = (_Float16)acc[r];
                if (rr < 64) sm.qk[row][h * 64 + rr] = hv;
                else         sm.vts[h][rr - 64][row] = hv;
            }
        }
    }
    __syncthreads();

    // ---- Stage C: attention, 2 heads in flight, 1 wave per 16-row strip ----
    const float scale = 0.17677669529663687f; // 32^-0.5
    const int mt = w & 3;    // row strip
    const int g  = w >> 2;   // head slot (0/1)

    // hoist all relative-position index math out of the head loop:
    // rows depend only on (mt, halfSel, r); cols only on (nt, laneN)
    int  bidx[4][8];
    bool mok[4][8];
    {
        int ih[8], iw[8];
        bool iok[8];
#pragma unroll
        for (int r = 0; r < 8; ++r) {
            int i = mt * 16 + halfSel * 8 + r;
            iok[r] = (i < NTOK);
            ih[r] = i / RES;
            iw[r] = i % RES;
        }
#pragma unroll
        for (int nt = 0; nt < 4; ++nt) {
            int j = nt * 16 + laneN;
            bool jok = (j < NTOK);
            int jh = j / RES, jw = j % RES;
#pragma unroll
            for (int r = 0; r < 8; ++r) {
                int di = ih[r] - jh; if (di < 0) di = -di;
                int dj = iw[r] - jw; if (dj < 0) dj = -dj;
                bidx[nt][r] = di * RES + dj;
                mok[nt][r]  = iok[r] && jok;
            }
        }
    }

    for (int hp = 0; hp < 4; ++hp) {
        const int h = hp * 2 + g;
        // S = Q K^T : K-dim = 32 -> exactly one wmma per 16x16 tile
        v16h qf = *(const v16h*)&sm.qk[mt * 16 + laneN][h * 64 + halfSel * 16];
        float pv[4][8];
#pragma unroll
        for (int nt = 0; nt < 4; ++nt) {
            v16h kf = *(const v16h*)&sm.qk[nt * 16 + laneN][h * 64 + 32 + halfSel * 16];
            v8f sacc;
#pragma unroll
            for (int i = 0; i < 8; ++i) sacc[i] = 0.f;
            sacc = __builtin_amdgcn_wmma_f32_16x16x32_f16(
                false, qf, false, kf, (short)0, sacc, false, false);
#pragma unroll
            for (int r = 0; r < 8; ++r) {
                pv[nt][r] = mok[nt][r]
                              ? sacc[r] * scale + sm.absm[h][bidx[nt][r]]
                              : -1e30f;   // mask padded rows/cols
            }
        }
        // row softmax: each row lives in one 16-lane half -> xor-shuffle reduce
#pragma unroll
        for (int r = 0; r < 8; ++r) {
            float mx = pv[0][r];
#pragma unroll
            for (int nt = 1; nt < 4; ++nt) mx = fmaxf(mx, pv[nt][r]);
            for (int off = 1; off < 16; off <<= 1) mx = fmaxf(mx, __shfl_xor(mx, off, 32));
            float ssum = 0.f;
#pragma unroll
            for (int nt = 0; nt < 4; ++nt) {
                float e = __expf(pv[nt][r] - mx);
                pv[nt][r] = e;
                ssum += e;
            }
            for (int off = 1; off < 16; off <<= 1) ssum += __shfl_xor(ssum, off, 32);
            float inv = 1.0f / ssum;
#pragma unroll
            for (int nt = 0; nt < 4; ++nt)
                sm.u.c.p[g][mt * 16 + halfSel * 8 + r][nt * 16 + laneN] =
                    (_Float16)(pv[nt][r] * inv);
        }
        // O = P @ V : per-wave LDS round-trip (C-layout -> A-layout transpose)
#pragma unroll
        for (int on = 0; on < 8; ++on) {
            v8f oacc;
#pragma unroll
            for (int i = 0; i < 8; ++i) oacc[i] = 0.f;
#pragma unroll
            for (int kk = 0; kk < 2; ++kk) {
                v16h pf = *(const v16h*)&sm.u.c.p[g][mt * 16 + laneN][kk * 32 + halfSel * 16];
                v16h vf = *(const v16h*)&sm.vts[h][on * 16 + laneN][kk * 32 + halfSel * 16];
                oacc = __builtin_amdgcn_wmma_f32_16x16x32_f16(
                    false, pf, false, vf, (short)0, oacc, false, false);
            }
#pragma unroll
            for (int r = 0; r < 8; ++r) {
                int i = mt * 16 + halfSel * 8 + r;
                if (i < NTOK)
                    sm.u.c.o[i][h * VAL_DIM + on * 16 + laneN] = (_Float16)oacc[r];
            }
        }
    }
    __syncthreads();

    // ---- Stage D: out = O @ Wproj + bias  (M=64, N=384, K=1024) ------------
    v16h zf;
#pragma unroll
    for (int i = 0; i < 16; ++i) zf[i] = (_Float16)0.f;
    for (int nn = 0; nn < 3; ++nn) {
        const int nt = w + nn * 8;                 // 24 n-tiles over 8 waves
        const float pb = proj_bias[nt * 16 + laneN];
        v8f acc[4];
#pragma unroll
        for (int m2 = 0; m2 < 4; ++m2)
#pragma unroll
            for (int i = 0; i < 8; ++i) acc[m2][i] = pb;
        const _Float16* wc = projT + (size_t)(nt * 16 + laneN) * 1024 + halfSel * 16;
        // warm caches for this wave's next n-tile
        __builtin_prefetch(wc + (size_t)8 * 16 * 1024, 0, 0);
        for (int k = 0; k < KSTEPS_PROJ; ++k) {
            v16h bf = *(const v16h*)(wc + k * 32);
#pragma unroll
            for (int m2 = 0; m2 < 4; ++m2) {
                int row = m2 * 16 + laneN;
                v16h af = (row < NTOK)
                            ? *(const v16h*)&sm.u.c.o[row][k * 32 + halfSel * 16]
                            : zf;
                acc[m2] = __builtin_amdgcn_wmma_f32_16x16x32_f16(
                    false, af, false, bf, (short)0, acc[m2], false, false);
            }
        }
        float* ob = out + (size_t)b * NTOK * DMODEL + nt * 16 + laneN;
#pragma unroll
        for (int m2 = 0; m2 < 4; ++m2)
#pragma unroll
            for (int r = 0; r < 8; ++r) {
                int i = m2 * 16 + halfSel * 8 + r;
                if (i < NTOK) ob[(size_t)i * DMODEL] = acc[m2][r];
            }
    }
}

extern "C" void kernel_launch(void* const* d_in, const int* in_sizes, int n_in,
                              void* d_out, int out_size, void* d_ws, size_t ws_size,
                              hipStream_t stream) {
    const float* x     = (const float*)d_in[0];
    const float* qkvK  = (const float*)d_in[1];
    const float* qkvB  = (const float*)d_in[2];
    const float* projK = (const float*)d_in[3];
    const float* projB = (const float*)d_in[4];
    const float* ab    = (const float*)d_in[5];
    float* out = (float*)d_out;

    // ws: f16 transposed weights (1.92 MB) -> L2-resident across all 2048 WGs
    _Float16* qkvT  = (_Float16*)d_ws;
    _Float16* projT = qkvT + (size_t)QKV_OUT * DMODEL;

    convert_weights<<<512, 256, 0, stream>>>(qkvK, projK, qkvT, projT);

    const size_t shmem = sizeof(SharedMem); // ~315 KB (<= 320 KB WGP LDS)
    hipFuncSetAttribute(reinterpret_cast<const void*>(attn4d_fused),
                        hipFuncAttributeMaxDynamicSharedMemorySize, (int)shmem);
    attn4d_fused<<<2048, 256, shmem, stream>>>(x, qkvT, qkvB, projT, projB, ab, out);
}
